// RNNAttention_46059229282562
// MI455X (gfx1250) — matrix operand order repaired and checked by reading, not compiled, not run on hardware
//
#include <hip/hip_runtime.h>
#include <hip/hip_bf16.h>
#include <math.h>

// Problem constants (from reference)
#define BB   8
#define SS   2048
#define EE   256
#define HH   256
#define VV   100
#define NROW (BB * SS)   // 16384 rows of [B*S, ...] activations
#define G3H  (3 * HH)    // 768

typedef __bf16 bf16;
typedef __attribute__((ext_vector_type(16))) __bf16 v16bf;
typedef __attribute__((ext_vector_type(8)))  float  v8f;

static __device__ __forceinline__ v8f wmma_bf16(v16bf a, v16bf b, v8f c) {
  // (neg_a, A, neg_b, B, c_mod, C, reuse_a, reuse_b)
  return __builtin_amdgcn_wmma_f32_16x16x32_bf16(false, a, false, b, (short)0, c,
                                                 false, false);
}

static __device__ __forceinline__ v8f vzero8() {
  v8f z = {0.f, 0.f, 0.f, 0.f, 0.f, 0.f, 0.f, 0.f};
  return z;
}

// ---- CDNA5 WMMA tile loaders (16x16x32 bf16 shapes) ---------------------
// Per-lane base pointers are computed once; k-steps advance by +32 elements.

// A (16x32, MxK): lanes 0-15 -> row=lane, K={k..k+7, k+16..k+23};
// lanes 16-31 -> row=lane-16, K={k+8..k+15, k+24..k+31}.
static __device__ __forceinline__ const bf16* a_lane_ptr(const bf16* X, int ld,
                                                         int mBase, int kBase) {
  int lane = threadIdx.x & 31;
  int row  = mBase + (lane & 15);
  int k0   = kBase + ((lane >> 4) << 3);   // 0 or 8
  return X + (size_t)row * ld + k0;
}
static __device__ __forceinline__ v16bf ld_a(const bf16* p) {
  union { v16bf v; uint4 u[2]; } t;
  t.u[0] = *reinterpret_cast<const uint4*>(p);        // k0   .. k0+7
  t.u[1] = *reinterpret_cast<const uint4*>(p + 16);   // k0+16.. k0+23
  return t.v;
}

// B (32x16, KxN) from row-major W[N,K]: lanes 0-15 -> col=lane, K={k..k+7, k+16..k+23};
// lanes 16-31 -> col=lane-16, K={k+8..k+15, k+24..k+31}.
static __device__ __forceinline__ const bf16* b_lane_ptr(const bf16* W, int ld,
                                                         int nBase, int kBase) {
  int lane = threadIdx.x & 31;
  int n  = nBase + (lane & 15);
  int k0 = kBase + ((lane >> 4) << 3);     // 0 or 8
  return W + (size_t)n * ld + k0;
}
static __device__ __forceinline__ v16bf ld_b(const bf16* p) {
  union { v16bf v; uint4 u[2]; } t;
  t.u[0] = *reinterpret_cast<const uint4*>(p);
  t.u[1] = *reinterpret_cast<const uint4*>(p + 16);
  return t.v;
}

// ---- small utility kernels ----------------------------------------------
__global__ void cvt_f32_bf16_kernel(const float* __restrict__ src,
                                    bf16* __restrict__ dst, int n) {
  int i = blockIdx.x * blockDim.x + threadIdx.x;
  if (i < n) dst[i] = (bf16)src[i];
}

__global__ void fill_zero_bf16_kernel(bf16* __restrict__ dst, int n) {
  int i = blockIdx.x * blockDim.x + threadIdx.x;
  if (i < n) dst[i] = (bf16)0.0f;
}

// Embedding gather straight to bf16: one thread per 8 elements.
__global__ void embed_gather_kernel(const int* __restrict__ x,
                                    const float* __restrict__ table,
                                    bf16* __restrict__ emb) {
  int i   = blockIdx.x * blockDim.x + threadIdx.x;
  int row = i >> 5;         // 32 chunks of 8 per 256-wide row
  int ch  = i & 31;
  if (row >= NROW) return;
  int tok = x[row];
  const float* src = table + (size_t)tok * EE + ch * 8;
  union { bf16 d[8]; uint4 u; } t;
#pragma unroll
  for (int k = 0; k < 8; ++k) t.d[k] = (bf16)src[k];
  *reinterpret_cast<uint4*>(emb + (size_t)row * EE + ch * 8) = t.u;
}

// Copy all_h into cols [0,256) of the concat buffer (cols [256,512) from flash).
__global__ void concat_copy_kernel(const bf16* __restrict__ allh,
                                   bf16* __restrict__ concat) {
  int i   = blockIdx.x * blockDim.x + threadIdx.x;
  int row = i >> 5;
  int ch  = i & 31;
  if (row >= NROW) return;
  *reinterpret_cast<uint4*>(concat + (size_t)row * (2 * HH) + ch * 8) =
      *reinterpret_cast<const uint4*>(allh + (size_t)row * HH + ch * 8);
}

// Transpose all_h [B*S, H] -> vT [B, H, S] (for contiguous P@V B-tiles).
__global__ void transpose_allh_kernel(const bf16* __restrict__ in,
                                      bf16* __restrict__ out) {
  __shared__ bf16 tile[32][33];
  int b  = blockIdx.z;
  int s0 = blockIdx.x * 32;
  int h0 = blockIdx.y * 32;
  int tx = threadIdx.x, ty = threadIdx.y;   // 32 x 8
#pragma unroll
  for (int i = 0; i < 32; i += 8)
    tile[ty + i][tx] = in[((size_t)b * SS + s0 + ty + i) * HH + h0 + tx];
  __syncthreads();
#pragma unroll
  for (int i = 0; i < 32; i += 8)
    out[((size_t)b * HH + h0 + ty + i) * SS + s0 + tx] = tile[tx][ty + i];
}

// ---- generic WMMA GEMM: C[M,N] = A[M,K](bf16) * W[N,K]^T(bf16) + bias ---
// Each wave computes a 16x64 strip; k-loop is register double-buffered so
// next-step tile loads are in flight while the current WMMAs execute.
__global__ void gemm_bf16_kernel(const bf16* __restrict__ A,
                                 const bf16* __restrict__ W,
                                 const float* __restrict__ bias,
                                 float* __restrict__ outF,
                                 bf16* __restrict__ outB,
                                 int M, int N, int K, int ldc, int nBound,
                                 int doTanh) {
  int wave   = blockIdx.x * (blockDim.x >> 5) + (threadIdx.x >> 5);
  int tilesN = N >> 6;
  int total  = (M >> 4) * tilesN;
  if (wave >= total) return;
  int m = (wave / tilesN) << 4;
  int n = (wave % tilesN) << 6;

  v8f acc[4];
#pragma unroll
  for (int j = 0; j < 4; ++j) acc[j] = vzero8();

  const bf16* pa = a_lane_ptr(A, K, m, 0);
  const bf16* pb[4];
#pragma unroll
  for (int j = 0; j < 4; ++j) pb[j] = b_lane_ptr(W, K, n + (j << 4), 0);

  v16bf a_cur = ld_a(pa);
  v16bf b_cur[4];
#pragma unroll
  for (int j = 0; j < 4; ++j) b_cur[j] = ld_b(pb[j]);

  for (int kk = 0; kk + 32 < K; kk += 32) {
    pa += 32;
    v16bf a_nxt = ld_a(pa);
    __builtin_prefetch(pa + 32, 0, 1);   // global_prefetch_b8 for k+2 strip
    v16bf b_nxt[4];
#pragma unroll
    for (int j = 0; j < 4; ++j) { pb[j] += 32; b_nxt[j] = ld_b(pb[j]); }
#pragma unroll
    for (int j = 0; j < 4; ++j) acc[j] = wmma_bf16(a_cur, b_cur[j], acc[j]);
    a_cur = a_nxt;
#pragma unroll
    for (int j = 0; j < 4; ++j) b_cur[j] = b_nxt[j];
  }
#pragma unroll
  for (int j = 0; j < 4; ++j) acc[j] = wmma_bf16(a_cur, b_cur[j], acc[j]);

  int lane  = threadIdx.x & 31;
  int rbase = m + ((lane >> 4) << 3);
#pragma unroll
  for (int j = 0; j < 4; ++j) {
    int col = n + (j << 4) + (lane & 15);
    if (col >= nBound) continue;
    float bv = bias ? bias[col] : 0.0f;
#pragma unroll
    for (int r = 0; r < 8; ++r) {
      float v = acc[j][r] + bv;
      if (doTanh) v = tanhf(v);
      size_t off = (size_t)(rbase + r) * ldc + col;
      if (outF) outF[off] = v;
      else      outB[off] = (bf16)v;
    }
  }
}

// ---- GRU scan: one persistent workgroup, h in LDS, 48 WMMAs per step ----
__global__ void gru_scan_kernel(const float* __restrict__ xg,
                                const bf16* __restrict__ whh,
                                const float* __restrict__ bhh,
                                bf16* __restrict__ allh,
                                float* __restrict__ hlast) {
  __shared__ bf16  h_s[16][HH];     // rows 8..15 stay zero (M padding)
  __shared__ float hg_s[BB][G3H];
  int t = threadIdx.x;
  for (int i = t; i < 16 * HH; i += 256) (&h_s[0][0])[i] = (bf16)0.0f;
  __syncthreads();

  int wv   = t >> 5;
  int lane = t & 31;
  int rb   = (lane >> 4) << 3;

  for (int s = 0; s < SS; ++s) {
    // hg = h @ w_hh^T  (16x768, K=256) split over 8 waves x 6 N-tiles
    v8f acc[6];
#pragma unroll
    for (int j = 0; j < 6; ++j) acc[j] = vzero8();

    const bf16* pa = a_lane_ptr(&h_s[0][0], HH, 0, 0);
    const bf16* pb[6];
#pragma unroll
    for (int j = 0; j < 6; ++j) pb[j] = b_lane_ptr(whh, HH, (wv * 6 + j) << 4, 0);

    v16bf a_cur = ld_a(pa);
    v16bf b_cur[6];
#pragma unroll
    for (int j = 0; j < 6; ++j) b_cur[j] = ld_b(pb[j]);

    for (int kk = 0; kk + 32 < HH; kk += 32) {
      pa += 32;
      v16bf a_nxt = ld_a(pa);
      v16bf b_nxt[6];
#pragma unroll
      for (int j = 0; j < 6; ++j) { pb[j] += 32; b_nxt[j] = ld_b(pb[j]); }
#pragma unroll
      for (int j = 0; j < 6; ++j) acc[j] = wmma_bf16(a_cur, b_cur[j], acc[j]);
      a_cur = a_nxt;
#pragma unroll
      for (int j = 0; j < 6; ++j) b_cur[j] = b_nxt[j];
    }
#pragma unroll
    for (int j = 0; j < 6; ++j) acc[j] = wmma_bf16(a_cur, b_cur[j], acc[j]);

#pragma unroll
    for (int j = 0; j < 6; ++j) {
      int col = ((wv * 6 + j) << 4) + (lane & 15);
#pragma unroll
      for (int r = 0; r < 8; ++r) {
        int row = rb + r;
        if (row < BB) hg_s[row][col] = acc[j][r];
      }
    }
    __syncthreads();

    // gate math: thread t owns hidden column t for all 8 batch rows
    int col = t;
#pragma unroll
    for (int b = 0; b < BB; ++b) {
      size_t xo = ((size_t)b * SS + s) * G3H;
      float xr = xg[xo + col];
      float xz = xg[xo + HH + col];
      float xn = xg[xo + 2 * HH + col];
      float hr = hg_s[b][col] + bhh[col];
      float hz = hg_s[b][HH + col] + bhh[HH + col];
      float hn = hg_s[b][2 * HH + col] + bhh[2 * HH + col];
      float r  = 1.0f / (1.0f + expf(-(xr + hr)));
      float z  = 1.0f / (1.0f + expf(-(xz + hz)));
      float nn = tanhf(xn + r * hn);
      float hp = (float)h_s[b][col];
      float hv = (1.0f - z) * nn + z * hp;
      h_s[b][col] = (bf16)hv;
      allh[((size_t)b * SS + s) * HH + col] = (bf16)hv;
      if (s == SS - 1) hlast[b * HH + col] = hv;
    }
    __syncthreads();
  }
}

// ---- causal flash attention: one wave per 16-query tile -----------------
__global__ void flash_attn_kernel(const bf16* __restrict__ allh,
                                  const bf16* __restrict__ keys,
                                  const bf16* __restrict__ vT,
                                  bf16* __restrict__ concat) {
  __shared__ bf16 pbuf[8][16 * 32];   // wave-private P bounce (D->A layout)
  int lane = threadIdx.x & 31;
  int wv   = threadIdx.x >> 5;
  int gw   = blockIdx.x * (blockDim.x >> 5) + wv;
  int b    = gw >> 7;                 // 128 query tiles per batch
  int qt   = gw & 127;
  int qBase = qt << 4;

  const bf16* Q  = allh + (size_t)b * SS * HH;
  const bf16* Kp = keys + (size_t)b * SS * HH;
  const bf16* V  = vT   + (size_t)b * HH * SS;

  v16bf qa[8];
  {
    const bf16* pq = a_lane_ptr(Q, HH, qBase, 0);
#pragma unroll
    for (int kk = 0; kk < 8; ++kk) { qa[kk] = ld_a(pq); pq += 32; }
  }

  v8f ctx[16];
#pragma unroll
  for (int j = 0; j < 16; ++j) ctx[j] = vzero8();
  float mrow[8], lrow[8];
#pragma unroll
  for (int r = 0; r < 8; ++r) { mrow[r] = -1e30f; lrow[r] = 0.0f; }

  int rb = (lane >> 4) << 3;
  int cn = lane & 15;
  int nChunks = (qBase >> 5) + 1;     // causal: kv chunks of 32 up to q

  for (int c = 0; c < nChunks; ++c) {
    int kv = c << 5;

    // scores: two 16x16 tiles, K pipelined (b tiles for kk+1 prefetched)
    v8f s0 = vzero8(), s1 = vzero8();
    {
      const bf16* pk = b_lane_ptr(Kp, HH, kv, 0);         // tile (kv,   *)
      const bf16* pk1 = pk + (size_t)16 * HH;             // tile (kv+16,*)
      v16bf b0c = ld_b(pk), b1c = ld_b(pk1);
#pragma unroll
      for (int kk = 0; kk < 7; ++kk) {
        pk += 32; pk1 += 32;
        v16bf b0n = ld_b(pk), b1n = ld_b(pk1);
        s0 = wmma_bf16(qa[kk], b0c, s0);
        s1 = wmma_bf16(qa[kk], b1c, s1);
        b0c = b0n; b1c = b1n;
      }
      s0 = wmma_bf16(qa[7], b0c, s0);
      s1 = wmma_bf16(qa[7], b1c, s1);
    }

    // online softmax update (row stats via half-wave shuffle reductions)
    float p0[8], p1[8], scale[8];
#pragma unroll
    for (int r = 0; r < 8; ++r) {
      int qrow = qBase + rb + r;
      float v0 = (kv + cn      <= qrow) ? s0[r] : -1e30f;
      float v1 = (kv + 16 + cn <= qrow) ? s1[r] : -1e30f;
      float mx = fmaxf(v0, v1);
      mx = fmaxf(mx, __shfl_xor(mx, 1));
      mx = fmaxf(mx, __shfl_xor(mx, 2));
      mx = fmaxf(mx, __shfl_xor(mx, 4));
      mx = fmaxf(mx, __shfl_xor(mx, 8));
      float mnew = fmaxf(mrow[r], mx);
      scale[r] = expf(mrow[r] - mnew);
      p0[r] = expf(v0 - mnew);
      p1[r] = expf(v1 - mnew);
      float ps = p0[r] + p1[r];
      ps += __shfl_xor(ps, 1);
      ps += __shfl_xor(ps, 2);
      ps += __shfl_xor(ps, 4);
      ps += __shfl_xor(ps, 8);
      lrow[r] = lrow[r] * scale[r] + ps;
      mrow[r] = mnew;
    }
#pragma unroll
    for (int j = 0; j < 16; ++j)
#pragma unroll
      for (int r = 0; r < 8; ++r) ctx[j][r] *= scale[r];

    // P (D layout) -> LDS -> A layout
    bf16* pb = &pbuf[wv][0];
#pragma unroll
    for (int r = 0; r < 8; ++r) {
      pb[(rb + r) * 32 + cn]      = (bf16)p0[r];
      pb[(rb + r) * 32 + 16 + cn] = (bf16)p1[r];
    }
    v16bf pa = ld_a(a_lane_ptr(pb, 32, 0, 0));

    // ctx += P(16x32) @ V(32x256): 16 N-tiles, pipelined over j
    {
      const bf16* pv = b_lane_ptr(V, SS, 0, kv);
      v16bf bc = ld_b(pv);
#pragma unroll
      for (int j = 0; j < 15; ++j) {
        pv += (size_t)16 * SS;
        v16bf bn = ld_b(pv);
        ctx[j] = wmma_bf16(pa, bc, ctx[j]);
        bc = bn;
      }
      ctx[15] = wmma_bf16(pa, bc, ctx[15]);
    }
  }

  float inv[8];
#pragma unroll
  for (int r = 0; r < 8; ++r) inv[r] = 1.0f / lrow[r];
#pragma unroll
  for (int j = 0; j < 16; ++j) {
    int col = HH + (j << 4) + cn;
#pragma unroll
    for (int r = 0; r < 8; ++r) {
      size_t row = (size_t)b * SS + qBase + rb + r;
      concat[row * (2 * HH) + col] = (bf16)(ctx[j][r] * inv[r]);
    }
  }
}

// ---- host-side orchestration --------------------------------------------
extern "C" void kernel_launch(void* const* d_in, const int* in_sizes, int n_in,
                              void* d_out, int out_size, void* d_ws, size_t ws_size,
                              hipStream_t stream) {
  const int*   x      = (const int*)  d_in[0];
  const float* embT   = (const float*)d_in[1];
  const float* w_ih   = (const float*)d_in[2];
  const float* w_hh   = (const float*)d_in[3];
  const float* b_ih   = (const float*)d_in[4];
  const float* b_hh   = (const float*)d_in[5];
  const float* attn_w = (const float*)d_in[6];
  const float* attn_b = (const float*)d_in[7];
  const float* comb_w = (const float*)d_in[8];
  const float* comb_b = (const float*)d_in[9];
  const float* fc_w   = (const float*)d_in[10];
  const float* fc_b   = (const float*)d_in[11];
  (void)in_sizes; (void)n_in; (void)out_size; (void)ws_size;

  char*  ws  = (char*)d_ws;
  size_t off = 0;
  auto alloc = [&](size_t bytes) -> void* {
    void* p = ws + off;
    off = (off + bytes + 255) & ~(size_t)255;
    return p;
  };
  bf16* emb_b   = (bf16*)alloc((size_t)NROW * EE * 2);     // embedded tokens
  bf16* wih_b   = (bf16*)alloc((size_t)G3H * EE * 2);
  bf16* whh_b   = (bf16*)alloc((size_t)G3H * HH * 2);
  bf16* attw_b  = (bf16*)alloc((size_t)HH * HH * 2);
  bf16* combw_b = (bf16*)alloc((size_t)HH * 2 * HH * 2);
  bf16* fcw_b   = (bf16*)alloc((size_t)128 * HH * 2);      // padded 100 -> 128
  float* xg     = (float*)alloc((size_t)NROW * G3H * 4);
  bf16* allh_b  = (bf16*)alloc((size_t)NROW * HH * 2);
  bf16* allhT_b = (bf16*)alloc((size_t)BB * HH * SS * 2);
  bf16* keys_b  = (bf16*)alloc((size_t)NROW * HH * 2);
  bf16* concat_b= (bf16*)alloc((size_t)NROW * 2 * HH * 2);
  bf16* comb_bh = (bf16*)alloc((size_t)NROW * HH * 2);

  float* out_logits = (float*)d_out;                      // [B,S,100]
  float* out_hlast  = (float*)d_out + (size_t)NROW * VV;  // [1,B,256]

  const int T = 256;
  auto blks = [](int n, int t) { return (n + t - 1) / t; };

  // 1) weight conversions (fc padded with zeros first)
  fill_zero_bf16_kernel<<<blks(128 * HH, T), T, 0, stream>>>(fcw_b, 128 * HH);
  cvt_f32_bf16_kernel<<<blks(G3H * EE, T), T, 0, stream>>>(w_ih, wih_b, G3H * EE);
  cvt_f32_bf16_kernel<<<blks(G3H * HH, T), T, 0, stream>>>(w_hh, whh_b, G3H * HH);
  cvt_f32_bf16_kernel<<<blks(HH * HH, T), T, 0, stream>>>(attn_w, attw_b, HH * HH);
  cvt_f32_bf16_kernel<<<blks(HH * 2 * HH, T), T, 0, stream>>>(comb_w, combw_b, HH * 2 * HH);
  cvt_f32_bf16_kernel<<<blks(VV * HH, T), T, 0, stream>>>(fc_w, fcw_b, VV * HH);

  // 2) embedding gather -> bf16
  embed_gather_kernel<<<blks(NROW * 32, T), T, 0, stream>>>(x, embT, emb_b);

  // 3) xg = emb @ w_ih^T + b_ih   [16384 x 768], f32 out
  {
    int waves = (NROW / 16) * (G3H / 64);
    gemm_bf16_kernel<<<waves / 8, T, 0, stream>>>(
        emb_b, wih_b, b_ih, xg, nullptr, NROW, G3H, EE, G3H, G3H, 0);
  }

  // 4) GRU scan (persistent single workgroup); also writes h_last
  gru_scan_kernel<<<1, T, 0, stream>>>(xg, whh_b, b_hh, allh_b, out_hlast);

  // 5) keys = all_h @ attn_w^T + attn_b  -> bf16
  {
    int waves = (NROW / 16) * (HH / 64);
    gemm_bf16_kernel<<<waves / 8, T, 0, stream>>>(
        allh_b, attw_b, attn_b, nullptr, keys_b, NROW, HH, HH, HH, HH, 0);
  }

  // 6) V^T for contiguous P@V tiles; concat cols [0,256) = all_h
  transpose_allh_kernel<<<dim3(SS / 32, HH / 32, BB), dim3(32, 8), 0, stream>>>(
      allh_b, allhT_b);
  concat_copy_kernel<<<blks(NROW * 32, T), T, 0, stream>>>(allh_b, concat_b);

  // 7) causal flash attention -> concat cols [256,512)
  flash_attn_kernel<<<(BB * (SS / 16)) / 8, T, 0, stream>>>(
      allh_b, keys_b, allhT_b, concat_b);

  // 8) combined = tanh(concat @ comb_w^T + comb_b) -> bf16
  {
    int waves = (NROW / 16) * (HH / 64);
    gemm_bf16_kernel<<<waves / 8, T, 0, stream>>>(
        concat_b, combw_b, comb_b, nullptr, comb_bh, NROW, HH, 2 * HH, HH, HH, 1);
  }

  // 9) logits = combined @ fc_w^T + fc_b (N padded to 128, store cols < 100)
  {
    int waves = (NROW / 16) * (128 / 64);
    gemm_bf16_kernel<<<waves / 8, T, 0, stream>>>(
        comb_bh, fcw_b, fc_b, out_logits, nullptr, NROW, 128, HH, VV, VV, 0);
  }
}
